// SwitchFFN_68478958567881
// MI455X (gfx1250) — matrix operand order repaired and checked by reading, not compiled, not run on hardware
//
#include <hip/hip_runtime.h>
#include <hip/hip_bf16.h>
#include <math.h>
#include <stdint.h>

// ---------------- problem constants (match reference) ----------------
#define DIM    1024
#define HIDDEN 4096
#define NEXP   8
#define NTOK   4096      // B*L = 2*2048

// ---------------- GEMM tiling ----------------
#define BM 128
#define BN 128
#define BK 32
#define LDSK 40          // bf16 elements per LDS row (stride); 40*2=80 bytes, 16B aligned

typedef __attribute__((ext_vector_type(16))) __bf16 v16bf;
typedef __attribute__((ext_vector_type(8)))  __bf16 v8bf;
typedef __attribute__((ext_vector_type(4)))  __bf16 v4bf;
typedef __attribute__((ext_vector_type(2)))  __bf16 v2bf;
typedef __attribute__((ext_vector_type(8)))  float  v8f;
typedef __attribute__((ext_vector_type(4)))  float  vf4;
typedef __attribute__((ext_vector_type(2)))  float  vf2;

union BF16x16 { v16bf v; v8bf h[2]; };
union F32x8   { v8f v; float f[8]; };

static __device__ __forceinline__ __bf16 f2bf(float f) { return (__bf16)f; }

// float2 -> packed bf16x2 dword; lowers to a single v_cvt_pk_bf16_f32
static __device__ __forceinline__ unsigned pk2bf(float lo, float hi) {
    vf2 s; s.x = lo; s.y = hi;
    v2bf d = __builtin_convertvector(s, v2bf);
    return __builtin_bit_cast(unsigned, d);
}

static __device__ __forceinline__ float gelu_exact(float v) {
    return 0.5f * v * (1.0f + erff(v * 0.70710678118654752f));
}

// generic -> LDS(addrspace 3) byte offset, for async-to-LDS instructions
typedef __attribute__((address_space(3))) char as3_char;
static __device__ __forceinline__ unsigned lds_byte_off(const void* p) {
    return (unsigned)(uintptr_t)(as3_char*)(p);
}

// A-fragment: 16x32 bf16 tile, rows striped across lanes per CDNA5 layout.
// lane<16 : M=lane,     K = {0..7, 16..23}
// lane>=16: M=lane-16,  K = {8..15, 24..31}
static __device__ __forceinline__ v16bf load_a_frag(const __bf16* lds, int rowBase, int lane) {
    int l = lane & 15, hi = lane >> 4;
    const __bf16* p = lds + (rowBase + l) * LDSK;
    BF16x16 f;
    f.h[0] = *(const v8bf*)(p + hi * 8);
    f.h[1] = *(const v8bf*)(p + 16 + hi * 8);
    return f.v;
}

// B-fragment: 32x16 bf16 (KxN), lds holds B transposed as [n][k].
// lane<16 : N=lane, K = {0..7, 16..23} ; lane>=16: N=lane-16, K = {8..15, 24..31}
static __device__ __forceinline__ v16bf load_b_frag(const __bf16* lds, int colBase, int lane) {
    int l = lane & 15, hi = lane >> 4;
    const __bf16* p = lds + (colBase + l) * LDSK + hi * 8;
    BF16x16 f;
    f.h[0] = *(const v8bf*)(p);
    f.h[1] = *(const v8bf*)(p + 16);
    return f.v;
}

// ------------------------------------------------------------------
// Kernel 0: zero the per-expert counters
// ------------------------------------------------------------------
__global__ void moe_zero_counts(int* counts) {
    if (threadIdx.x < NEXP) counts[threadIdx.x] = 0;
}

// ------------------------------------------------------------------
// Kernel 1: router. One wave (32 lanes) per token. Wg cached in LDS.
// ------------------------------------------------------------------
__global__ __launch_bounds__(256) void moe_router(const float* __restrict__ x,
                                                  const float* __restrict__ Wg,
                                                  int* __restrict__ counts,
                                                  int* __restrict__ lists) {
    __shared__ float sWg[NEXP * DIM];   // 32 KB
    int tid  = threadIdx.x;
    int lane = tid & 31;
    int wave = tid >> 5;

    for (int i = tid; i < NEXP * DIM; i += 256) sWg[i] = Wg[i];
    __syncthreads();

    int token = blockIdx.x * 8 + wave;              // grid = 512 blocks -> 4096 tokens
    const float* xr = x + (size_t)token * DIM;

    float acc[NEXP];
#pragma unroll
    for (int e = 0; e < NEXP; ++e) acc[e] = 0.0f;

    for (int d = lane; d < DIM; d += 32) {
        float xv = xr[d];
#pragma unroll
        for (int e = 0; e < NEXP; ++e) acc[e] += xv * sWg[e * DIM + d];
    }

    // wave32 tree reduction
#pragma unroll
    for (int off = 16; off >= 1; off >>= 1) {
#pragma unroll
        for (int e = 0; e < NEXP; ++e) acc[e] += __shfl_xor(acc[e], off, 32);
    }

    if (lane == 0) {
        int   best = 0;
        float bv   = acc[0];
#pragma unroll
        for (int e = 1; e < NEXP; ++e) {
            if (acc[e] > bv) { bv = acc[e]; best = e; }   // first-max tie-break, like argmax
        }
        int pos = atomicAdd(&counts[best], 1);
        lists[best * NTOK + pos] = token;
    }
}

// ------------------------------------------------------------------
// Kernel 2: GEMM1  H[token, :] = gelu(x[token, :] @ W1[e] + b1[e])
// grid = (HIDDEN/BN=32, NTOK/BM=32, NEXP=8), block = 256 (8 waves)
// ------------------------------------------------------------------
__global__ __launch_bounds__(256) void moe_gemm1(const float* __restrict__ x,
                                                 const float* __restrict__ W1,
                                                 const float* __restrict__ b1,
                                                 const int* __restrict__ counts,
                                                 const int* __restrict__ lists,
                                                 __bf16* __restrict__ Hbuf) {
    int e   = blockIdx.z;
    int cnt = counts[e];
    int m0  = blockIdx.y * BM;
    if (m0 >= cnt) return;                       // uniform: EXEC stays all-ones for WMMA
    int n0  = blockIdx.x * BN;

    __shared__ __attribute__((aligned(16))) __bf16 sA[BM * LDSK];
    __shared__ __attribute__((aligned(16))) __bf16 sB[BN * LDSK];
    __shared__ int sTok[BM];

    int tid  = threadIdx.x;
    int lane = tid & 31;
    int wave = tid >> 5;
    int wm   = wave >> 1;   // 0..3 : 32-row strip
    int wn   = wave & 1;    // 0..1 : 64-col strip

    const int*   list = lists + e * NTOK;
    const float* W    = W1 + (size_t)e * DIM * HIDDEN;
    const float* bias = b1 + (size_t)e * HIDDEN;

    for (int i = tid; i < BM; i += 256) {
        int m = m0 + i;
        // rows past cnt: invalid for the epilogue; staging clamps to sTok[0] = list[m0]
        sTok[i] = (m < cnt) ? list[m] : -1;
    }
    __syncthreads();

    // ---- hoist loop-invariant staging addresses out of the k-loop ----
    const float* aSrc[4];   // A: 1024 float4 chunks / 256 threads
    int          aOff[4];
#pragma unroll
    for (int i = 0; i < 4; ++i) {
        int c   = tid + 256 * i;
        int row = c >> 3;            // 8 float4 per 32-wide row
        int kq  = c & 7;
        int tk  = sTok[row];
        int tks = tk < 0 ? sTok[0] : tk;
        aSrc[i] = x + (size_t)tks * DIM + kq * 4;
        aOff[i] = row * LDSK + kq * 4;
    }
    const float* bSrc[2];   // B: 512 chunks (two k-rows each) / 256 threads
    int          bOff[2];
#pragma unroll
    for (int i = 0; i < 2; ++i) {
        int c  = tid + 256 * i;
        int kp = c >> 5;             // k-pair 0..15
        int nq = c & 31;             // n-quad 0..31
        int kk = kp * 2;
        bSrc[i] = W + (size_t)kk * HIDDEN + n0 + nq * 4;
        bOff[i] = (nq * 4) * LDSK + kk;
    }

    v8f acc[2][4];
#pragma unroll
    for (int mi = 0; mi < 2; ++mi)
#pragma unroll
        for (int ni = 0; ni < 4; ++ni) acc[mi][ni] = (v8f)(0.0f);

    for (int k0 = 0; k0 < DIM; k0 += BK) {
        // ---- stage A: f32 -> bf16x4, one 8B packed store per chunk
#pragma unroll
        for (int i = 0; i < 4; ++i) {
            vf4 xv = *(const vf4*)(aSrc[i] + k0);
            *(v4bf*)(sA + aOff[i]) = __builtin_convertvector(xv, v4bf);
        }
        // ---- stage B: two k-rows per thread, packed k-pairs -> 4B LDS stores
#pragma unroll
        for (int i = 0; i < 2; ++i) {
            const float* p0 = bSrc[i] + (size_t)k0 * HIDDEN;
            vf4 w0 = *(const vf4*)p0;
            vf4 w1 = *(const vf4*)(p0 + HIDDEN);
            __bf16* d = sB + bOff[i];
            *(unsigned*)(d + 0 * LDSK) = pk2bf(w0.x, w1.x);
            *(unsigned*)(d + 1 * LDSK) = pk2bf(w0.y, w1.y);
            *(unsigned*)(d + 2 * LDSK) = pk2bf(w0.z, w1.z);
            *(unsigned*)(d + 3 * LDSK) = pk2bf(w0.w, w1.w);
        }
        __syncthreads();

        v16bf aF[2];
#pragma unroll
        for (int mi = 0; mi < 2; ++mi)
            aF[mi] = load_a_frag(sA, wm * 32 + mi * 16, lane);
#pragma unroll
        for (int ni = 0; ni < 4; ++ni) {
            v16bf bF = load_b_frag(sB, wn * 64 + ni * 16, lane);
#pragma unroll
            for (int mi = 0; mi < 2; ++mi) {
                acc[mi][ni] = __builtin_amdgcn_wmma_f32_16x16x32_bf16(
                    false, aF[mi], false, bF, (short)0, acc[mi][ni], false, false);
            }
        }
        __syncthreads();
    }

    // ---- epilogue: bias + exact GELU -> bf16 H
    int l  = lane & 15;
    int hi = lane >> 4;
#pragma unroll
    for (int ni = 0; ni < 4; ++ni) {
        int col = n0 + wn * 64 + ni * 16 + l;
        float bv = bias[col];
#pragma unroll
        for (int mi = 0; mi < 2; ++mi) {
            F32x8 u; u.v = acc[mi][ni];
#pragma unroll
            for (int r = 0; r < 8; ++r) {
                int mrow = wm * 32 + mi * 16 + hi * 8 + r;
                int tk = sTok[mrow];
                if (tk >= 0) {
                    float v = gelu_exact(u.f[r] + bv);
                    Hbuf[(size_t)tk * HIDDEN + col] = f2bf(v);
                }
            }
        }
    }
}

// ------------------------------------------------------------------
// Kernel 3: GEMM2  out[token, :] = H[token, :] @ W2[e] + b2[e]
// grid = (DIM/BN=8, NTOK/BM=32, NEXP=8), block = 256
// A tile is already bf16 -> stage via async global->LDS (ASYNCcnt path)
// ------------------------------------------------------------------
__global__ __launch_bounds__(256) void moe_gemm2(const __bf16* __restrict__ Hbuf,
                                                 const float* __restrict__ W2,
                                                 const float* __restrict__ b2,
                                                 const int* __restrict__ counts,
                                                 const int* __restrict__ lists,
                                                 float* __restrict__ out) {
    int e   = blockIdx.z;
    int cnt = counts[e];
    int m0  = blockIdx.y * BM;
    if (m0 >= cnt) return;
    int n0  = blockIdx.x * BN;

    __shared__ __attribute__((aligned(16))) __bf16 sA[BM * LDSK];
    __shared__ __attribute__((aligned(16))) __bf16 sB[BN * LDSK];
    __shared__ int sTok[BM];

    int tid  = threadIdx.x;
    int lane = tid & 31;
    int wave = tid >> 5;
    int wm   = wave >> 1;
    int wn   = wave & 1;

    const int*   list = lists + e * NTOK;
    const float* W    = W2 + (size_t)e * HIDDEN * DIM;
    const float* bias = b2 + (size_t)e * DIM;

    for (int i = tid; i < BM; i += 256) {
        int m = m0 + i;
        sTok[i] = (m < cnt) ? list[m] : -1;
    }
    __syncthreads();

    // ---- hoist staging addresses ----
    const __bf16* gSrc[2];   // A: 512 b128 chunks / 256 threads
    unsigned      gLds[2];
#pragma unroll
    for (int i = 0; i < 2; ++i) {
        int c   = tid + 256 * i;
        int row = c >> 2;            // 4 chunks per 32-wide row
        int kq  = c & 3;
        int tk  = sTok[row];
        int tks = tk < 0 ? sTok[0] : tk;
        gSrc[i] = Hbuf + (size_t)tks * HIDDEN + kq * 8;
        gLds[i] = lds_byte_off(sA + row * LDSK + kq * 8);
    }
    const float* bSrc[2];
    int          bOff[2];
#pragma unroll
    for (int i = 0; i < 2; ++i) {
        int c  = tid + 256 * i;
        int kp = c >> 5;
        int nq = c & 31;
        int kk = kp * 2;
        bSrc[i] = W + (size_t)kk * DIM + n0 + nq * 4;
        bOff[i] = (nq * 4) * LDSK + kk;
    }

    v8f acc[2][4];
#pragma unroll
    for (int mi = 0; mi < 2; ++mi)
#pragma unroll
        for (int ni = 0; ni < 4; ++ni) acc[mi][ni] = (v8f)(0.0f);

    for (int k0 = 0; k0 < HIDDEN; k0 += BK) {
        // ---- stage A: bf16 H rows, async global->LDS b128 (ASYNCcnt-tracked)
#pragma unroll
        for (int i = 0; i < 2; ++i) {
            asm volatile("global_load_async_to_lds_b128 %0, %1, off"
                         :: "v"(gLds[i]),
                            "v"((unsigned long long)(uintptr_t)(gSrc[i] + k0))
                         : "memory");
        }
        // ---- stage B: W2 slice, two k-rows per thread, packed 4B LDS stores
#pragma unroll
        for (int i = 0; i < 2; ++i) {
            const float* p0 = bSrc[i] + (size_t)k0 * DIM;
            vf4 w0 = *(const vf4*)p0;
            vf4 w1 = *(const vf4*)(p0 + DIM);
            __bf16* d = sB + bOff[i];
            *(unsigned*)(d + 0 * LDSK) = pk2bf(w0.x, w1.x);
            *(unsigned*)(d + 1 * LDSK) = pk2bf(w0.y, w1.y);
            *(unsigned*)(d + 2 * LDSK) = pk2bf(w0.z, w1.z);
            *(unsigned*)(d + 3 * LDSK) = pk2bf(w0.w, w1.w);
        }
        asm volatile("s_wait_asynccnt 0x0" ::: "memory");    // A tile landed in LDS
        __syncthreads();

        v16bf aF[2];
#pragma unroll
        for (int mi = 0; mi < 2; ++mi)
            aF[mi] = load_a_frag(sA, wm * 32 + mi * 16, lane);
#pragma unroll
        for (int ni = 0; ni < 4; ++ni) {
            v16bf bF = load_b_frag(sB, wn * 64 + ni * 16, lane);
#pragma unroll
            for (int mi = 0; mi < 2; ++mi) {
                acc[mi][ni] = __builtin_amdgcn_wmma_f32_16x16x32_bf16(
                    false, aF[mi], false, bF, (short)0, acc[mi][ni], false, false);
            }
        }
        __syncthreads();
    }

    // ---- epilogue: + bias, scatter f32 rows to output (every token covered once)
    int l  = lane & 15;
    int hi = lane >> 4;
#pragma unroll
    for (int ni = 0; ni < 4; ++ni) {
        int col = n0 + wn * 64 + ni * 16 + l;
        float bv = bias[col];
#pragma unroll
        for (int mi = 0; mi < 2; ++mi) {
            F32x8 u; u.v = acc[mi][ni];
#pragma unroll
            for (int r = 0; r < 8; ++r) {
                int mrow = wm * 32 + mi * 16 + hi * 8 + r;
                int tk = sTok[mrow];
                if (tk >= 0) out[(size_t)tk * DIM + col] = u.f[r] + bv;
            }
        }
    }
}

// ------------------------------------------------------------------
// Host-side launcher
// ------------------------------------------------------------------
extern "C" void kernel_launch(void* const* d_in, const int* in_sizes, int n_in,
                              void* d_out, int out_size, void* d_ws, size_t ws_size,
                              hipStream_t stream) {
    const float* x  = (const float*)d_in[0];
    const float* Wg = (const float*)d_in[1];
    const float* W1 = (const float*)d_in[2];
    const float* b1 = (const float*)d_in[3];
    const float* W2 = (const float*)d_in[4];
    const float* b2 = (const float*)d_in[5];
    float* out = (float*)d_out;

    char* ws = (char*)d_ws;
    int*    counts = (int*)ws;                                  // 8 ints
    int*    lists  = (int*)(ws + 256);                          // 8*4096 ints = 128 KB
    __bf16* Hbuf   = (__bf16*)(ws + 256 * 1024);                // 4096*4096 bf16 = 33.5 MB

    moe_zero_counts<<<1, 32, 0, stream>>>(counts);
    moe_router<<<NTOK / 8, 256, 0, stream>>>(x, Wg, counts, lists);
    moe_gemm1<<<dim3(HIDDEN / BN, NTOK / BM, NEXP), 256, 0, stream>>>(
        x, W1, b1, counts, lists, Hbuf);
    moe_gemm2<<<dim3(DIM / BN, NTOK / BM, NEXP), 256, 0, stream>>>(
        Hbuf, W2, b2, counts, lists, out);
}
